// LocFeacoding_37855841747269
// MI455X (gfx1250) — matrix operand order repaired and checked by reading, not compiled, not run on hardware
//
#include <hip/hip_runtime.h>
#include <math.h>

// ---------- types ----------
typedef __attribute__((ext_vector_type(4)))  unsigned v4u;
typedef __attribute__((ext_vector_type(4)))  float    v4f;
typedef __attribute__((ext_vector_type(8)))  float    v8f;
typedef __attribute__((ext_vector_type(16))) __bf16   v16bf;

union BPack { v4u u[2]; v16bf v; };

__device__ __forceinline__ unsigned short f2bf(float f) {
  unsigned u = __builtin_bit_cast(unsigned, f);
  u += 0x7fffu + ((u >> 16) & 1u);           // round-to-nearest-even
  return (unsigned short)(u >> 16);
}

// Problem constants
#define BB   4
#define NN   4096
#define KK   32
#define CIN  256
#define SC   128
#define COUT 512
#define MM   16384   // B*N samples

// ---------------------------------------------------------------------------
// Weight bf16 conversion + zero stats
// ---------------------------------------------------------------------------
__global__ void k_prep(const float* __restrict__ wsub, const float* __restrict__ wgcm,
                       const float* __restrict__ wup,  const float* __restrict__ wres,
                       unsigned short* wsub_bf, unsigned short* wgcm_bf,
                       unsigned short* wup_bf,  unsigned short* wres_bf,
                       float* stats) {
  int i = blockIdx.x * blockDim.x + threadIdx.x;
  if (i < SC * CIN)   wsub_bf[i] = f2bf(wsub[i]);     // 32768
  if (i < SC * 2*SC)  wgcm_bf[i] = f2bf(wgcm[i]);     // 32768
  if (i < COUT * SC)  wup_bf[i]  = f2bf(wup[i]);      // 65536
  if (i < COUT * CIN) wres_bf[i] = f2bf(wres[i]);     // 131072
  if (i < 2560)       stats[i] = 0.f;
}

// feature (B,256,N) f32 -> feat_bf (B*N, 256) bf16
__global__ void k_feat_t(const float* __restrict__ feat, unsigned short* __restrict__ out) {
  for (int i = blockIdx.x * blockDim.x + threadIdx.x; i < MM * CIN;
       i += gridDim.x * blockDim.x) {
    int c = i & 255, m = i >> 8;
    int b = m >> 12, n = m & 4095;
    out[i] = f2bf(feat[((size_t)b * CIN + c) * NN + n]);
  }
}

// ---------------------------------------------------------------------------
// Generic bf16 WMMA GEMM: A(M,K) row-major, Bm(N,K) row-major,
// output stored TRANSPOSED: Ct[n*MM + m] (channel-major, f32).
// grid: (MM/16, Ndim/128), block 256 (8 waves, one 16x16 n-subtile each)
// ---------------------------------------------------------------------------
__global__ void k_gemm(const unsigned short* __restrict__ A,
                       const unsigned short* __restrict__ Bm,
                       float* __restrict__ Ct, int Kdim) {
  int lane = threadIdx.x & 31, wave = threadIdx.x >> 5;
  int m0 = blockIdx.x * 16;
  int n0 = blockIdx.y * 128 + wave * 16;
  int am    = m0 + (lane & 15);
  int akoff = (lane < 16) ? 0 : 8;
  int bn    = n0 + (lane & 15);
  int bkoff = (lane < 16) ? 0 : 8;
  v8f acc = {};
  for (int k0 = 0; k0 < Kdim; k0 += 32) {
    BPack a, b;
    const v4u* pa = (const v4u*)(A + (size_t)am * Kdim + k0 + akoff);
    a.u[0] = pa[0]; a.u[1] = pa[2];                    // K: +0..7 and +16..23
    const v4u* pb = (const v4u*)(Bm + (size_t)bn * Kdim + k0 + bkoff);
    b.u[0] = pb[0]; b.u[1] = pb[2];                    // K: +0..7 and +16..23
    acc = __builtin_amdgcn_wmma_f32_16x16x32_bf16(false, a.v, false, b.v,
                                                  (short)0, acc, false, false);
  }
  int col   = n0 + (lane & 15);
  int rbase = m0 + ((lane < 16) ? 0 : 8);
  v4f lo = { acc[0], acc[1], acc[2], acc[3] };
  v4f hi = { acc[4], acc[5], acc[6], acc[7] };
  *(v4f*)(Ct + (size_t)col * MM + rbase)     = lo;
  *(v4f*)(Ct + (size_t)col * MM + rbase + 4) = hi;
}

// per-channel sum/sumsq over a contiguous row of length MM (deterministic)
__global__ void k_rowstats(const float* __restrict__ src, float* sumo, float* sqo) {
  __shared__ float s1[256], s2[256];
  int c = blockIdx.x;
  const float* row = src + (size_t)c * MM;
  float s = 0.f, q = 0.f;
  for (int i = threadIdx.x; i < MM; i += 256) { float v = row[i]; s += v; q += v * v; }
  s1[threadIdx.x] = s; s2[threadIdx.x] = q; __syncthreads();
  for (int off = 128; off; off >>= 1) {
    if (threadIdx.x < (unsigned)off) { s1[threadIdx.x] += s1[threadIdx.x + off];
                                       s2[threadIdx.x] += s2[threadIdx.x + off]; }
    __syncthreads();
  }
  if (threadIdx.x == 0) { sumo[c] = s1[0]; sqo[c] = s2[0]; }
}

// BN fold: a = g*rsqrt(var+eps), d = be - a*mean   (conv bias cancels in BN)
__global__ void k_finstats(const float* sum, const float* sq, const float* g,
                           const float* be, float invN, float* a, float* d, int C) {
  int c = blockIdx.x * blockDim.x + threadIdx.x;
  if (c < C) {
    float m = sum[c] * invN;
    float v = sq[c] * invN - m * m;
    float ac = g[c] * rsqrtf(v + 1e-5f);
    a[c] = ac; d[c] = be[c] - ac * m;
  }
}

// sub_rawT (128, MM) -> sub_bf (MM, 128) bf16 with BN+ReLU
__global__ void k_sub_apply(const float* __restrict__ subT, const float* __restrict__ a,
                            const float* __restrict__ d, unsigned short* __restrict__ out) {
  for (int i = blockIdx.x * blockDim.x + threadIdx.x; i < MM * SC;
       i += gridDim.x * blockDim.x) {
    int c = i & 127, m = i >> 7;
    float v = fmaxf(a[c] * subT[(size_t)c * MM + m] + d[c], 0.f);
    out[i] = f2bf(v);
  }
}

// ---------------------------------------------------------------------------
// Fused gcm kernel. One block handles (b, n0..n0+1): 128 out-channels x 64 cols
// (2 n * 32 k), K=256 (128 exp + 128 gathered sub). Two passes (template).
// Pass1: deterministic per-block partial stats. Pass2: BN+ReLU + attention
// score + softmax over channels + k-max-pool -> pooled bf16.
// Gathered-sub half of the X tile is moved with GLOBAL_LOAD_ASYNC_TO_LDS_B128
// (ASYNCcnt path; no VGPR staging), exp half is b128-vectorized + converted.
// ---------------------------------------------------------------------------
#define XST 264   // padded bf16 stride of X tile rows (cols); XST*2 = 528 = 33*16
#define YST 68    // padded f32 stride of y rows (channels)

template <bool PASS2>
__global__ void k_gcm(const float* __restrict__ exp_f, const int* __restrict__ gidx,
                      const unsigned short* __restrict__ sub_bf,
                      const unsigned short* __restrict__ wg_bf,
                      const float* __restrict__ watt, const float* __restrict__ batt,
                      const float* __restrict__ a2, const float* __restrict__ d2,
                      float* __restrict__ part, unsigned short* __restrict__ pooled) {
  __shared__ alignas(16) unsigned char smem[36864];
  unsigned short* Xs = (unsigned short*)smem;        // 64 x XST bf16 (33792 B)
  float* Ys = (float*)smem;                          // 128 x YST f32 (34816 B), aliases Xs
  float* Ss = (float*)(smem + 34816);                // 256 f32 scores

  int b = blockIdx.y, n0 = blockIdx.x * 2, t = threadIdx.x;

  // gathered-sub half via async global->LDS b128 (16B per lane-op, 16 per col)
  for (int j = t; j < 64 * 16; j += 256) {
    int col = j >> 4, part = j & 15, nl = col >> 5, k = col & 31;
    int idx = gidx[((size_t)b * NN + n0 + nl) * KK + k];
    const unsigned short* g = sub_bf + ((size_t)b * NN + idx) * SC + part * 8;
    unsigned lds = (unsigned)(uintptr_t)(Xs + col * XST + SC + part * 8);
    asm volatile("global_load_async_to_lds_b128 %0, %1, off"
                 :: "v"(lds), "v"(g) : "memory");
  }

  // exp half: X[col][c] = expanded_f[b,c,n0+nl,k],  col = nl*32+k (vectorized x4)
  for (int i = t; i < SC * 16; i += 256) {
    int c = i >> 4, col0 = (i & 15) * 4;
    const v4f* pe = (const v4f*)(exp_f + (((size_t)b * SC + c) * NN + n0) * KK + col0);
    v4f v = *pe;
    Xs[(col0 + 0) * XST + c] = f2bf(v[0]);
    Xs[(col0 + 1) * XST + c] = f2bf(v[1]);
    Xs[(col0 + 2) * XST + c] = f2bf(v[2]);
    Xs[(col0 + 3) * XST + c] = f2bf(v[3]);
  }

  asm volatile("s_wait_asynccnt 0" ::: "memory");
  __syncthreads();

  // GEMM: 8 waves x (16 o) each covering 4 col-tiles, K=256
  int lane = t & 31, wave = t >> 5;
  int o0    = wave * 16;
  int am    = o0 + (lane & 15);
  int akoff = (lane < 16) ? 0 : 8;
  int bkoff = (lane < 16) ? 0 : 8;
  v8f acc[4] = {};
  for (int k0 = 0; k0 < 2 * SC; k0 += 32) {
    BPack a;
    const v4u* pa = (const v4u*)(wg_bf + (size_t)am * (2 * SC) + k0 + akoff);
    a.u[0] = pa[0]; a.u[1] = pa[2];
#pragma unroll
    for (int ct = 0; ct < 4; ct++) {
      BPack bb;
      const v4u* pb = (const v4u*)(Xs + (ct * 16 + (lane & 15)) * XST + k0 + bkoff);
      bb.u[0] = pb[0]; bb.u[1] = pb[2];
      acc[ct] = __builtin_amdgcn_wmma_f32_16x16x32_bf16(false, a.v, false, bb.v,
                                                        (short)0, acc[ct], false, false);
    }
  }
  __syncthreads();                    // X fully consumed -> safe to alias with Ys
  int rbase = o0 + ((lane < 16) ? 0 : 8);
#pragma unroll
  for (int ct = 0; ct < 4; ct++) {
    int col = ct * 16 + (lane & 15);
#pragma unroll
    for (int r = 0; r < 8; r++) Ys[(rbase + r) * YST + col] = acc[ct][r];
  }
  __syncthreads();

  if (!PASS2) {
    if (t < SC) {
      float s = 0.f, q = 0.f;
      for (int col = 0; col < 64; col++) { float v = Ys[t * YST + col]; s += v; q += v * v; }
      int pb = blockIdx.y * 2048 + blockIdx.x;
      part[(size_t)pb * 256 + t]      = s;
      part[(size_t)pb * 256 + SC + t] = q;
    }
  } else {
    int nl = t >> 7, c = t & 127;
    float ac = a2[c], dc = d2[c];
    float sc = batt[0], mx = -1e30f;
#pragma unroll
    for (int k = 0; k < KK; k++) {
      float v = fmaxf(ac * Ys[c * YST + nl * 32 + k] + dc, 0.f);
      sc += watt[k] * v;
      mx = fmaxf(mx, v);
    }
    Ss[t] = sc;
    __syncthreads();
    int base = nl * 128;
    float smax = -1e30f;
    for (int j = 0; j < 128; j++) smax = fmaxf(smax, Ss[base + j]);
    float se = 0.f;
    for (int j = 0; j < 128; j++) se += __expf(Ss[base + j] - smax);
    float p = mx * (1.f + __expf(sc - smax) / se);
    pooled[((size_t)b * NN + n0 + nl) * SC + c] = f2bf(p);
  }
}

// fixed-order reduction of gcm partial stats (deterministic)
__global__ void k_redstats(const float* __restrict__ part, float* sum, float* sq) {
  __shared__ float s1[256], s2[256];
  int c = blockIdx.x;
  float s = 0.f, q = 0.f;
  for (int i = threadIdx.x; i < 8192; i += 256) {
    s += part[(size_t)i * 256 + c];
    q += part[(size_t)i * 256 + SC + c];
  }
  s1[threadIdx.x] = s; s2[threadIdx.x] = q; __syncthreads();
  for (int off = 128; off; off >>= 1) {
    if (threadIdx.x < (unsigned)off) { s1[threadIdx.x] += s1[threadIdx.x + off];
                                       s2[threadIdx.x] += s2[threadIdx.x + off]; }
    __syncthreads();
  }
  if (threadIdx.x == 0) { sum[c] = s1[0]; sq[c] = s2[0]; }
}

// out[b,o,n] = relu(a_up*upT + d_up) + relu(a_res*resT + d_res)
__global__ void k_final(const float* __restrict__ upT, const float* __restrict__ resT,
                        const float* __restrict__ au, const float* __restrict__ du,
                        const float* __restrict__ ar, const float* __restrict__ dr,
                        float* __restrict__ out) {
  for (int i = blockIdx.x * blockDim.x + threadIdx.x; i < BB * COUT * NN;
       i += gridDim.x * blockDim.x) {
    int n = i & 4095, o = (i >> 12) & 511, b = i >> 21;
    size_t m = (size_t)b * NN + n;
    float u = fmaxf(au[o] * upT[(size_t)o * MM + m] + du[o], 0.f);
    float r = fmaxf(ar[o] * resT[(size_t)o * MM + m] + dr[o], 0.f);
    out[i] = u + r;
  }
}

// ---------------------------------------------------------------------------
extern "C" void kernel_launch(void* const* d_in, const int* in_sizes, int n_in,
                              void* d_out, int out_size, void* d_ws, size_t ws_size,
                              hipStream_t stream) {
  const float* expanded = (const float*)d_in[0];
  const float* feature  = (const float*)d_in[1];
  const int*   gidx     = (const int*)d_in[2];
  const float* w_sub = (const float*)d_in[3];
  const float* g_sub = (const float*)d_in[5];
  const float* be_sub= (const float*)d_in[6];
  const float* w_gcm = (const float*)d_in[7];
  const float* g_gcm = (const float*)d_in[9];
  const float* be_gcm= (const float*)d_in[10];
  const float* w_att = (const float*)d_in[11];
  const float* b_att = (const float*)d_in[12];
  const float* w_up  = (const float*)d_in[13];
  const float* g_up  = (const float*)d_in[15];
  const float* be_up = (const float*)d_in[16];
  const float* w_res = (const float*)d_in[17];
  const float* g_res = (const float*)d_in[19];
  const float* be_res= (const float*)d_in[20];
  float* out = (float*)d_out;

  char* w = (char*)d_ws;
  unsigned short* wsub_bf = (unsigned short*)(w + 0);          //  64 KB
  unsigned short* wgcm_bf = (unsigned short*)(w + 65536);      //  64 KB
  unsigned short* wup_bf  = (unsigned short*)(w + 131072);     // 128 KB
  unsigned short* wres_bf = (unsigned short*)(w + 262144);     // 256 KB
  unsigned short* feat_bf = (unsigned short*)(w + 524288);     //   8 MB
  float*          subT    = (float*)(w + 8912896);             //   8 MB
  unsigned short* sub_bf  = (unsigned short*)(w + 17301504);   //   4 MB
  unsigned short* pooled  = (unsigned short*)(w + 21495808);   //   4 MB
  float*          upT     = (float*)(w + 25690112);            //  32 MB (pass1 partials alias)
  float*          resT    = (float*)(w + 59244544);            //  32 MB
  float*          stats   = (float*)(w + 92798976);
  float* sum_sub = stats,        *sq_sub = stats + 128;
  float* sum_gcm = stats + 256,  *sq_gcm = stats + 384;
  float* sum_up  = stats + 512,  *sq_up  = stats + 1024;
  float* sum_res = stats + 1536, *sq_res = stats + 2048;
  float* a_sub = stats + 2560, *d_sub = stats + 2688;
  float* a_gcm = stats + 2816, *d_gcm = stats + 2944;
  float* a_up  = stats + 3072, *d_up  = stats + 3584;
  float* a_res = stats + 4096, *d_res = stats + 4608;

  k_prep<<<512, 256, 0, stream>>>(w_sub, w_gcm, w_up, w_res,
                                  wsub_bf, wgcm_bf, wup_bf, wres_bf, stats);
  k_feat_t<<<4096, 256, 0, stream>>>(feature, feat_bf);

  // sub = ReLU(BN(w_sub @ feature)) -> bf16 (b,n,c)
  k_gemm<<<dim3(MM / 16, 1), 256, 0, stream>>>(feat_bf, wsub_bf, subT, CIN);
  k_rowstats<<<SC, 256, 0, stream>>>(subT, sum_sub, sq_sub);
  k_finstats<<<1, SC, 0, stream>>>(sum_sub, sq_sub, g_sub, be_sub,
                                   1.f / (float)MM, a_sub, d_sub, SC);
  k_sub_apply<<<2048, 256, 0, stream>>>(subT, a_sub, d_sub, sub_bf);

  // gcm pass1: raw stats (upT region used as scratch for partials)
  k_gcm<false><<<dim3(NN / 2, BB), 256, 0, stream>>>(expanded, gidx, sub_bf, wgcm_bf,
                                                     w_att, b_att, a_gcm, d_gcm,
                                                     upT, pooled);
  k_redstats<<<SC, 256, 0, stream>>>(upT, sum_gcm, sq_gcm);
  k_finstats<<<1, SC, 0, stream>>>(sum_gcm, sq_gcm, g_gcm, be_gcm,
                                   1.f / (float)(BB * NN * KK), a_gcm, d_gcm, SC);
  // gcm pass2: BN+ReLU + score/softmax + max-pool -> pooled
  k_gcm<true><<<dim3(NN / 2, BB), 256, 0, stream>>>(expanded, gidx, sub_bf, wgcm_bf,
                                                    w_att, b_att, a_gcm, d_gcm,
                                                    upT, pooled);

  // up / res GEMMs (transposed channel-major outputs)
  k_gemm<<<dim3(MM / 16, COUT / 128), 256, 0, stream>>>(pooled, wup_bf, upT, SC);
  k_gemm<<<dim3(MM / 16, COUT / 128), 256, 0, stream>>>(feat_bf, wres_bf, resT, CIN);
  k_rowstats<<<COUT, 256, 0, stream>>>(upT, sum_up, sq_up);
  k_rowstats<<<COUT, 256, 0, stream>>>(resT, sum_res, sq_res);
  k_finstats<<<2, 256, 0, stream>>>(sum_up, sq_up, g_up, be_up,
                                    1.f / (float)MM, a_up, d_up, COUT);
  k_finstats<<<2, 256, 0, stream>>>(sum_res, sq_res, g_res, be_res,
                                    1.f / (float)MM, a_res, d_res, COUT);

  k_final<<<8192, 256, 0, stream>>>(upT, resT, a_up, d_up, a_res, d_res, out);
}
